// PointNet2Encoder_64879775973813
// MI455X (gfx1250) — compile-verified, matrix operand order checked by reading, not run on hardware
//
#include <hip/hip_runtime.h>
#include <hip/hip_bf16.h>

#define BATCH 16
#define NPTS  4096
#define NP1   512
#define K1    32
#define NP2   128
#define K2    64
#define EPSV  1e-5f

typedef __attribute__((ext_vector_type(16))) _Float16 v16h;
typedef __attribute__((ext_vector_type(8)))  _Float16 v8h;
typedef __attribute__((ext_vector_type(8)))  float    v8f;

// ---------------------------------------------------------------------------
// Farthest point sampling: one workgroup per batch, min-dists in registers,
// LDS argmax tree each iteration (scan semantics: emit `far` BEFORE update).
// ---------------------------------------------------------------------------
__global__ void fps_kernel(const float* __restrict__ pts, int n, int npoint,
                           int* __restrict__ idxs) {
    const int b = blockIdx.x;
    const int t = threadIdx.x;
    const float* p = pts + (size_t)b * n * 3;
    __shared__ float sred[256];
    __shared__ int   sidx[256];
    __shared__ float scent[3];

    float dmin[16];
    const int cnt = (n + 255) / 256;
    for (int i = 0; i < cnt; ++i) dmin[i] = 1e10f;
    int far = 0;

    for (int it = 0; it < npoint; ++it) {
        if (t == 0) {
            idxs[b * npoint + it] = far;
            scent[0] = p[(size_t)far * 3 + 0];
            scent[1] = p[(size_t)far * 3 + 1];
            scent[2] = p[(size_t)far * 3 + 2];
        }
        __syncthreads();
        const float cx = scent[0], cy = scent[1], cz = scent[2];
        float best = -1.0f; int bi = 0;
        for (int i = 0; i < cnt; ++i) {
            int id = i * 256 + t;
            if (id < n) {
                float dx = p[(size_t)id * 3 + 0] - cx;
                float dy = p[(size_t)id * 3 + 1] - cy;
                float dz = p[(size_t)id * 3 + 2] - cz;
                float d = dx * dx + dy * dy + dz * dz;
                dmin[i] = fminf(dmin[i], d);
                if (dmin[i] > best) { best = dmin[i]; bi = id; }
            }
        }
        sred[t] = best; sidx[t] = bi;
        __syncthreads();
        for (int s = 128; s > 0; s >>= 1) {
            if (t < s) {
                if (sred[t + s] > sred[t]) { sred[t] = sred[t + s]; sidx[t] = sidx[t + s]; }
            }
            __syncthreads();
        }
        far = sidx[0];
        __syncthreads();
    }
}

// ---------------------------------------------------------------------------
__global__ void gather_centers_kernel(const float* __restrict__ pts,
                                      const int* __restrict__ idx,
                                      float* __restrict__ centers,
                                      int n, int npoint) {
    int i = blockIdx.x * blockDim.x + threadIdx.x;
    int total = BATCH * npoint * 3;
    if (i >= total) return;
    int c = i % 3; int j = (i / 3) % npoint; int b = i / (3 * npoint);
    int id = idx[b * npoint + j];
    centers[i] = pts[((size_t)b * n + id) * 3 + c];
}

// ---------------------------------------------------------------------------
// Ball query: one wave32 per center; ordered in-radius compaction via __ballot.
// ---------------------------------------------------------------------------
__global__ void ball_query_kernel(const float* __restrict__ pts,
                                  const float* __restrict__ centers,
                                  int n, int npoint, float r2, int k,
                                  int* __restrict__ out_idx, int totalWaves) {
    int gw = (blockIdx.x * blockDim.x + threadIdx.x) >> 5;
    int lane = threadIdx.x & 31;
    if (gw >= totalWaves) return;
    int b = gw / npoint, ctr = gw % npoint;
    const float* p = pts + (size_t)b * n * 3;
    const float* c = centers + ((size_t)b * npoint + ctr) * 3;
    float cx = c[0], cy = c[1], cz = c[2];
    int* dst = out_idx + ((size_t)b * npoint + ctr) * k;

    int cnt = 0;
    int first = n;
    for (int base = 0; base < n && cnt < k; base += 32) {
        int i = base + lane;
        bool ok = false;
        if (i < n) {
            float dx = p[(size_t)i * 3 + 0] - cx;
            float dy = p[(size_t)i * 3 + 1] - cy;
            float dz = p[(size_t)i * 3 + 2] - cz;
            ok = (dx * dx + dy * dy + dz * dz) <= r2;
        }
        unsigned mask = (unsigned)__ballot(ok);
        if (lane == 0) {
            while (mask && cnt < k) {
                int bit = __builtin_ctz(mask);
                mask &= mask - 1;
                int id = base + bit;
                if (first == n) first = id;
                dst[cnt++] = id;
            }
        }
        cnt = __shfl(cnt, 0);
    }
    if (lane == 0) {
        int fill = first;  // first in-radius index (center itself always hits)
        for (int j = cnt; j < k; ++j) dst[j] = fill;
    }
}

// ---------------------------------------------------------------------------
// Grouping kernels: build padded f16 A-operand activation matrices.
// ---------------------------------------------------------------------------
__global__ void group1_kernel(const float* __restrict__ x,
                              const float* __restrict__ centers1,
                              const int* __restrict__ idx1,
                              _Float16* __restrict__ A) {
    int row = blockIdx.x * blockDim.x + threadIdx.x;
    const int total = BATCH * NP1 * K1;
    if (row >= total) return;
    int g = row / K1; int ctr = g % NP1; int b = g / NP1;
    int id = idx1[row];
    float vx = 0.f, vy = 0.f, vz = 0.f;
    if (id < NPTS) {
        const float* p = x + ((size_t)b * NPTS + id) * 3;
        const float* c = centers1 + ((size_t)b * NP1 + ctr) * 3;
        vx = p[0] - c[0]; vy = p[1] - c[1]; vz = p[2] - c[2];
    }
    _Float16* d = A + (size_t)row * 32;
    d[0] = (_Float16)vx; d[1] = (_Float16)vy; d[2] = (_Float16)vz;
    for (int i = 3; i < 32; ++i) d[i] = (_Float16)0.f;
}

__global__ void group2_kernel(const float* __restrict__ centers1,
                              const float* __restrict__ centers2,
                              const int* __restrict__ idx2,
                              const _Float16* __restrict__ f1,
                              _Float16* __restrict__ A) {
    int row = blockIdx.x * blockDim.x + threadIdx.x;
    const int total = BATCH * NP2 * K2;
    if (row >= total) return;
    int g = row / K2; int ctr = g % NP2; int b = g / NP2;
    int id = idx2[row];
    _Float16* d = A + (size_t)row * 160;               // 3 + 128 padded to 160
    float vx = 0.f, vy = 0.f, vz = 0.f;
    const float* c = centers2 + ((size_t)b * NP2 + ctr) * 3;
    if (id < NP1) {
        const float* p = centers1 + ((size_t)b * NP1 + id) * 3;
        vx = p[0] - c[0]; vy = p[1] - c[1]; vz = p[2] - c[2];
    }
    d[0] = (_Float16)vx; d[1] = (_Float16)vy; d[2] = (_Float16)vz;
    const _Float16* f = f1 + ((size_t)b * NP1 + (id < NP1 ? id : 0)) * 128;
    for (int i = 0; i < 128; ++i) d[3 + i] = (id < NP1) ? f[i] : (_Float16)0.f;
    for (int i = 131; i < 160; ++i) d[i] = (_Float16)0.f;
}

__global__ void group3_kernel(const float* __restrict__ centers2,
                              const _Float16* __restrict__ f2,
                              _Float16* __restrict__ A) {
    int row = blockIdx.x * blockDim.x + threadIdx.x;
    const int total = BATCH * NP2;
    if (row >= total) return;
    const float* c = centers2 + (size_t)row * 3;
    _Float16* d = A + (size_t)row * 288;               // 3 + 256 padded to 288
    d[0] = (_Float16)c[0]; d[1] = (_Float16)c[1]; d[2] = (_Float16)c[2];
    const _Float16* f = f2 + (size_t)row * 256;
    for (int i = 0; i < 256; ++i) d[3 + i] = f[i];
    for (int i = 259; i < 288; ++i) d[i] = (_Float16)0.f;
}

// ---------------------------------------------------------------------------
// Weight pre-pack, WMMA B-operand lane layout:
//   Wp[((nt*nkb + kb)*32 + lane)*16 + j]
// where lane holds col = nt*16 + (lane&15) and its 16 halves are
// K { kb*32 + khalf + 0..7 , kb*32 + 16 + khalf + 0..7 }, khalf = (lane>>4)*8.
// This makes the GEMM's B fetch a single contiguous 32B load per lane.
// ---------------------------------------------------------------------------
__global__ void pack_w_kernel(const float* __restrict__ W, _Float16* __restrict__ Wp,
                              int cin, int cout, int kpad) {
    int idx = blockIdx.x * blockDim.x + threadIdx.x;
    int total = kpad * cout;
    if (idx >= total) return;
    int nkb  = kpad >> 5;
    int j    = idx & 15;
    int lane = (idx >> 4) & 31;
    int rest = idx >> 9;              // nt*nkb + kb
    int kb   = rest % nkb;
    int nt   = rest / nkb;
    int col   = nt * 16 + (lane & 15);
    int khalf = (lane >> 4) * 8;
    int k = kb * 32 + ((j < 8) ? (khalf + j) : (16 + khalf + (j - 8)));
    Wp[idx] = (k < cin) ? (_Float16)W[(size_t)k * cout + col] : (_Float16)0.f;
}

__global__ void zero_f32_kernel(float* __restrict__ p, int n) {
    int i = blockIdx.x * blockDim.x + threadIdx.x;
    if (i < n) p[i] = 0.f;
}

// ---------------------------------------------------------------------------
// WMMA GEMM, N-blocked x4: each wave owns a 16x64 output strip.
// Per K-step: one A fetch (2 x b128) reused by 4 v_wmma_f32_16x16x32_f16,
// B fetched contiguously from the packed layout (2 x b128 per tile).
// ---------------------------------------------------------------------------
__global__ void gemm_wmma_kernel(const _Float16* __restrict__ A,
                                 const _Float16* __restrict__ Wp,
                                 const float* __restrict__ bias,
                                 _Float16* __restrict__ Y,
                                 int M, int Kpad, int Cout, int totalWaves) {
    int gw = (blockIdx.x * blockDim.x + threadIdx.x) >> 5;
    int lane = threadIdx.x & 31;
    if (gw >= totalWaves) return;            // wave-uniform exit
    int ntg = Cout >> 6;                     // groups of 4 n-tiles (64 cols)
    int mt = gw / ntg, ng = gw % ntg;
    int nkb = Kpad >> 5;

    int row   = mt * 16 + (lane & 15);
    int khalf = (lane >> 4) * 8;

    v8f acc0 = {}, acc1 = {}, acc2 = {}, acc3 = {};
    const _Float16* arow = A + (size_t)row * Kpad;
    const size_t tileStride = (size_t)nkb * 512;       // halves per 16-col tile
    const _Float16* bbase = Wp + (size_t)(ng * 4) * tileStride;

    for (int kb = 0; kb < nkb; ++kb) {
        int k0 = kb * 32;
        v8h a_lo = *(const v8h*)(arow + k0 + khalf);
        v8h a_hi = *(const v8h*)(arow + k0 + 16 + khalf);
        v16h a;
        #pragma unroll
        for (int i = 0; i < 8; ++i) { a[i] = a_lo[i]; a[8 + i] = a_hi[i]; }

        const _Float16* bp = bbase + ((size_t)kb * 32 + lane) * 16;
        v16h b0 = *(const v16h*)(bp);
        v16h b1 = *(const v16h*)(bp + tileStride);
        v16h b2 = *(const v16h*)(bp + 2 * tileStride);
        v16h b3 = *(const v16h*)(bp + 3 * tileStride);

        acc0 = __builtin_amdgcn_wmma_f32_16x16x32_f16(false, a, false, b0, (short)0, acc0, false, false);
        acc1 = __builtin_amdgcn_wmma_f32_16x16x32_f16(false, a, false, b1, (short)0, acc1, false, false);
        acc2 = __builtin_amdgcn_wmma_f32_16x16x32_f16(false, a, false, b2, (short)0, acc2, false, false);
        acc3 = __builtin_amdgcn_wmma_f32_16x16x32_f16(false, a, false, b3, (short)0, acc3, false, false);
    }

    int rbase = mt * 16 + ((lane >> 4) << 3);
    v8f accs[4] = {acc0, acc1, acc2, acc3};
    #pragma unroll
    for (int t = 0; t < 4; ++t) {
        int col = ng * 64 + t * 16 + (lane & 15);
        float bv = bias[col];
        #pragma unroll
        for (int r = 0; r < 8; ++r)
            Y[(size_t)(rbase + r) * Cout + col] = (_Float16)(accs[t][r] + bv);
    }
}

// ---------------------------------------------------------------------------
// Per-channel sum / sum-of-squares (batchnorm over all M rows).
// ---------------------------------------------------------------------------
__global__ void stats_kernel(const _Float16* __restrict__ Y, float* __restrict__ stats,
                             int M, int C) {
    __shared__ float ss[256], sq[256];
    int c = blockIdx.x;
    int m0 = blockIdx.y * 8192;
    int mend = (m0 + 8192 < M) ? m0 + 8192 : M;
    float s = 0.f, q = 0.f;
    for (int m = m0 + threadIdx.x; m < mend; m += 256) {
        float v = (float)Y[(size_t)m * C + c];
        s += v; q += v * v;
    }
    ss[threadIdx.x] = s; sq[threadIdx.x] = q;
    __syncthreads();
    for (int st = 128; st > 0; st >>= 1) {
        if (threadIdx.x < st) {
            ss[threadIdx.x] += ss[threadIdx.x + st];
            sq[threadIdx.x] += sq[threadIdx.x + st];
        }
        __syncthreads();
    }
    if (threadIdx.x == 0) {
        atomicAdd(&stats[c], ss[0]);
        atomicAdd(&stats[C + c], sq[0]);
    }
}

__global__ void bn_relu_kernel(const _Float16* __restrict__ Y,
                               const float* __restrict__ stats,
                               const float* __restrict__ gamma,
                               const float* __restrict__ beta,
                               _Float16* __restrict__ Xn,
                               int M, int C, float invM) {
    size_t i = (size_t)blockIdx.x * blockDim.x + threadIdx.x;
    size_t total = (size_t)M * C;
    if (i >= total) return;
    int c = (int)(i % C);
    float mean = stats[c] * invM;
    float var  = stats[C + c] * invM - mean * mean;
    float v = ((float)Y[i] - mean) * rsqrtf(var + EPSV) * gamma[c] + beta[c];
    Xn[i] = (_Float16)fmaxf(v, 0.f);
}

// ---------------------------------------------------------------------------
// Max-pool over neighborhood (k rows per group, per channel).
// ---------------------------------------------------------------------------
__global__ void maxk_f16_kernel(const _Float16* __restrict__ X, _Float16* __restrict__ out,
                                int G, int k, int C) {
    int i = blockIdx.x * blockDim.x + threadIdx.x;
    if (i >= G * C) return;
    int c = i % C, g = i / C;
    const _Float16* p = X + (size_t)g * k * C + c;
    float m = -1e30f;
    for (int j = 0; j < k; ++j) m = fmaxf(m, (float)p[(size_t)j * C]);
    out[(size_t)g * C + c] = (_Float16)m;
}

__global__ void maxk_f32_kernel(const _Float16* __restrict__ X, float* __restrict__ out,
                                int G, int k, int C) {
    int i = blockIdx.x * blockDim.x + threadIdx.x;
    if (i >= G * C) return;
    int c = i % C, g = i / C;
    const _Float16* p = X + (size_t)g * k * C + c;
    float m = -1e30f;
    for (int j = 0; j < k; ++j) m = fmaxf(m, (float)p[(size_t)j * C]);
    out[(size_t)g * C + c] = m;
}

// ---------------------------------------------------------------------------
extern "C" void kernel_launch(void* const* d_in, const int* in_sizes, int n_in,
                              void* d_out, int out_size, void* d_ws, size_t ws_size,
                              hipStream_t stream) {
    (void)in_sizes; (void)n_in; (void)out_size; (void)ws_size;
    const float* x = (const float*)d_in[0];
    // params flattened in dict order: sa{1,2,3} x 3 layers x {W,b,gamma,beta}
    auto Wf  = [&](int s, int l) { return (const float*)d_in[1 + (s * 3 + l) * 4 + 0]; };
    auto Bf  = [&](int s, int l) { return (const float*)d_in[1 + (s * 3 + l) * 4 + 1]; };
    auto Gf  = [&](int s, int l) { return (const float*)d_in[1 + (s * 3 + l) * 4 + 2]; };
    auto Bef = [&](int s, int l) { return (const float*)d_in[1 + (s * 3 + l) * 4 + 3]; };

    // ----- workspace bump allocator (256B aligned) -----
    char* base = (char*)d_ws;
    size_t off = 0;
    auto alloc = [&](size_t bytes) -> void* {
        off = (off + 255) & ~(size_t)255;
        void* p = base + off;
        off += bytes;
        return p;
    };
    const size_t M1 = (size_t)BATCH * NP1 * K1;   // 262144
    const size_t M2 = (size_t)BATCH * NP2 * K2;   // 131072
    const size_t M3 = (size_t)BATCH * NP2;        // 2048

    _Float16* P     = (_Float16*)alloc(M1 * 128 * sizeof(_Float16)); // ping
    _Float16* Q     = (_Float16*)alloc(M1 * 128 * sizeof(_Float16)); // pong
    _Float16* Wp    = (_Float16*)alloc((size_t)512 * 1024 * sizeof(_Float16));
    float*    stats = (float*)   alloc(2 * 1024 * sizeof(float));
    int*   c1_idx   = (int*)     alloc((size_t)BATCH * NP1 * sizeof(int));
    float* centers1 = (float*)   alloc((size_t)BATCH * NP1 * 3 * sizeof(float));
    int*   idx1     = (int*)     alloc(M1 * sizeof(int));
    _Float16* f1    = (_Float16*)alloc((size_t)BATCH * NP1 * 128 * sizeof(_Float16));
    int*   c2_idx   = (int*)     alloc((size_t)BATCH * NP2 * sizeof(int));
    float* centers2 = (float*)   alloc((size_t)BATCH * NP2 * 3 * sizeof(float));
    int*   idx2     = (int*)     alloc(M2 * sizeof(int));
    _Float16* f2    = (_Float16*)alloc((size_t)BATCH * NP2 * 256 * sizeof(_Float16));

    auto cdiv = [](size_t a, size_t b) { return (unsigned)((a + b - 1) / b); };

    // Layer driver: pack weights, WMMA gemm (16x64 strips), stats, bn+relu.
    auto run_layer = [&](const _Float16* Ain, _Float16* Xout, int M, int Kpad,
                         int cin, int cout, int s, int l) {
        int wt = Kpad * cout;
        pack_w_kernel<<<cdiv(wt, 256), 256, 0, stream>>>(Wf(s, l), Wp, cin, cout, Kpad);
        int waves = (M / 16) * (cout / 64);
        gemm_wmma_kernel<<<cdiv((size_t)waves * 32, 256), 256, 0, stream>>>(
            Ain, Wp, Bf(s, l), Q, M, Kpad, cout, waves);
        zero_f32_kernel<<<cdiv(2 * cout, 256), 256, 0, stream>>>(stats, 2 * cout);
        dim3 sgrid(cout, cdiv(M, 8192));
        stats_kernel<<<sgrid, 256, 0, stream>>>(Q, stats, M, cout);
        size_t tot = (size_t)M * cout;
        bn_relu_kernel<<<cdiv(tot, 256), 256, 0, stream>>>(
            Q, stats, Gf(s, l), Bef(s, l), Xout, M, cout, 1.0f / (float)M);
    };

    // ===== Stage 1 =====
    fps_kernel<<<BATCH, 256, 0, stream>>>(x, NPTS, NP1, c1_idx);
    gather_centers_kernel<<<cdiv((size_t)BATCH * NP1 * 3, 256), 256, 0, stream>>>(
        x, c1_idx, centers1, NPTS, NP1);
    {
        int waves = BATCH * NP1;
        ball_query_kernel<<<cdiv((size_t)waves * 32, 256), 256, 0, stream>>>(
            x, centers1, NPTS, NP1, 0.2f * 0.2f, K1, idx1, waves);
    }
    group1_kernel<<<cdiv(M1, 256), 256, 0, stream>>>(x, centers1, idx1, P);
    run_layer(P, P, (int)M1, 32, 3, 64, 0, 0);
    run_layer(P, P, (int)M1, 64, 64, 64, 0, 1);
    run_layer(P, P, (int)M1, 64, 64, 128, 0, 2);
    maxk_f16_kernel<<<cdiv((size_t)BATCH * NP1 * 128, 256), 256, 0, stream>>>(
        P, f1, BATCH * NP1, K1, 128);

    // ===== Stage 2 =====
    fps_kernel<<<BATCH, 256, 0, stream>>>(centers1, NP1, NP2, c2_idx);
    gather_centers_kernel<<<cdiv((size_t)BATCH * NP2 * 3, 256), 256, 0, stream>>>(
        centers1, c2_idx, centers2, NP1, NP2);
    {
        int waves = BATCH * NP2;
        ball_query_kernel<<<cdiv((size_t)waves * 32, 256), 256, 0, stream>>>(
            centers1, centers2, NP1, NP2, 0.4f * 0.4f, K2, idx2, waves);
    }
    group2_kernel<<<cdiv(M2, 256), 256, 0, stream>>>(centers1, centers2, idx2, f1, P);
    run_layer(P, P, (int)M2, 160, 131, 128, 1, 0);
    run_layer(P, P, (int)M2, 128, 128, 128, 1, 1);
    run_layer(P, P, (int)M2, 128, 128, 256, 1, 2);
    maxk_f16_kernel<<<cdiv((size_t)BATCH * NP2 * 256, 256), 256, 0, stream>>>(
        P, f2, BATCH * NP2, K2, 256);

    // ===== Stage 3 =====
    group3_kernel<<<cdiv(M3, 256), 256, 0, stream>>>(centers2, f2, P);
    run_layer(P, P, (int)M3, 288, 259, 256, 2, 0);
    run_layer(P, P, (int)M3, 256, 256, 512, 2, 1);
    run_layer(P, P, (int)M3, 512, 512, 1024, 2, 2);
    maxk_f32_kernel<<<cdiv((size_t)BATCH * 1024, 256), 256, 0, stream>>>(
        P, (float*)d_out, BATCH, NP2, 1024);
}
